// RfcnPSRoIExtractor_41042707480840
// MI455X (gfx1250) — matrix operand order repaired and checked by reading, not compiled, not run on hardware
//
#include <hip/hip_runtime.h>
#include <hip/hip_bf16.h>

// ---------------------------------------------------------------------------
// RFCN PSROI extractor for gfx1250 (MI455X).
// Stage 0: pack fp32 operands into bf16 hi/lo pairs, pre-swizzled into the
//          exact v_wmma_f32_16x16x32_bf16 per-lane VGPR layout (one time).
// Stage 1: GEMM, one wave = 32(M) x 64(N) tile. Per K=32 slab:
//          8 A-loads + 16 B-loads (b128) -> 24 WMMAs (bf16 split:
//          hi*hi + hi*lo + lo*hi ~ fp32-class accuracy).
// Stage 2: position-sensitive RoI pooling (direct window averaging).
// ---------------------------------------------------------------------------

typedef __attribute__((ext_vector_type(16))) __bf16 v16bf;
typedef __attribute__((ext_vector_type(8)))  float  v8f;

#define KDIM 512
#define NHW  4200
#define FH   50
#define FW   84
#define PK   7
#define KT   16          // KDIM / 32 k-slabs
#define NT   263         // ceil(NHW / 16) column tiles

__device__ __forceinline__ unsigned short f2bf(float x) {
  unsigned int u = __float_as_uint(x);
  unsigned int r = (u + 0x7FFFu + ((u >> 16) & 1u)) >> 16;   // RNE to bf16
  return (unsigned short)r;
}
__device__ __forceinline__ float bf2f(unsigned short h) {
  return __uint_as_float(((unsigned int)h) << 16);
}

union U8 { unsigned int u[8]; uint4 q[2]; v16bf v; };

// ---------------------------------------------------------------------------
// Pack B (feats) into WMMA B-operand layout.
// Bhi/Blo layout: [b][kt][nt][lane(32)][8 x u32]
// lane l: column N = nt*16 + (l&15); u32 0..3 -> K=+0..7, u32 4..7 -> K=+16..23 (lanes 16-31: +8),
// two K values per u32 ([15:0] = even/lower K).
// ---------------------------------------------------------------------------
__global__ void pack_B(const float* __restrict__ F,   // [2, 512, 4200]
                       unsigned int* __restrict__ Bhi,
                       unsigned int* __restrict__ Blo)
{
  const int tid  = blockIdx.x * blockDim.x + threadIdx.x;
  const int lane = tid & 31;
  int t = tid >> 5;
  const int nt = t % NT;  t /= NT;
  const int kt = t % KT;
  const int b  = t / KT;
  if (b >= 2) return;

  const int l15     = lane & 15;
  const int halfSel = lane >> 4;
  const int col     = nt * 16 + l15;
  const int kb      = kt * 32 + halfSel * 8;

  U8 hi, lo;
  if (col < NHW) {
    const float* __restrict__ Fb = F + (size_t)b * KDIM * NHW + col;
    #pragma unroll
    for (int i = 0; i < 8; ++i) {
      const int kk = kb + ((i < 4) ? (2 * i) : (16 + 2 * (i - 4)));
      float x0 = Fb[(size_t)(kk)     * NHW];
      float x1 = Fb[(size_t)(kk + 1) * NHW];
      unsigned short h0 = f2bf(x0), h1 = f2bf(x1);
      unsigned short l0 = f2bf(x0 - bf2f(h0)), l1 = f2bf(x1 - bf2f(h1));
      hi.u[i] = (unsigned)h0 | ((unsigned)h1 << 16);
      lo.u[i] = (unsigned)l0 | ((unsigned)l1 << 16);
    }
  } else {
    #pragma unroll
    for (int i = 0; i < 8; ++i) { hi.u[i] = 0u; lo.u[i] = 0u; }
  }
  const size_t base = (size_t)tid * 8;
  ((uint4*)(Bhi + base))[0] = hi.q[0];
  ((uint4*)(Bhi + base))[1] = hi.q[1];
  ((uint4*)(Blo + base))[0] = lo.q[0];
  ((uint4*)(Blo + base))[1] = lo.q[1];
}

// ---------------------------------------------------------------------------
// Pack A (weights) into WMMA A-operand layout.
// Ahi/Alo layout: [mt][kt][lane(32)][8 x u32]
// lane holds row M = mt*16 + (lane&15);
// lanes 0-15: u32 0..3 -> K=+0..7, u32 4..7 -> K=+16..23; lanes 16-31: +8.
// ---------------------------------------------------------------------------
__global__ void pack_A(const float* __restrict__ W,   // [M, 512]
                       unsigned int* __restrict__ Ahi,
                       unsigned int* __restrict__ Alo,
                       int M, int MT)
{
  const int tid  = blockIdx.x * blockDim.x + threadIdx.x;
  const int lane = tid & 31;
  int t = tid >> 5;
  const int kt = t % KT;
  const int mt = t / KT;
  if (mt >= MT) return;

  const int l15     = lane & 15;
  const int halfSel = lane >> 4;
  const int row     = mt * 16 + l15;
  const int rowc    = row < M ? row : (M - 1);
  const int ka      = kt * 32 + halfSel * 8;
  const float* __restrict__ Wrow = W + (size_t)rowc * KDIM;

  U8 hi, lo;
  #pragma unroll
  for (int i = 0; i < 8; ++i) {
    const int base = ka + ((i < 4) ? (2 * i) : (16 + 2 * (i - 4)));
    float x0 = Wrow[base], x1 = Wrow[base + 1];
    unsigned short h0 = f2bf(x0), h1 = f2bf(x1);
    unsigned short l0 = f2bf(x0 - bf2f(h0)), l1 = f2bf(x1 - bf2f(h1));
    hi.u[i] = (unsigned)h0 | ((unsigned)h1 << 16);
    lo.u[i] = (unsigned)l0 | ((unsigned)l1 << 16);
  }
  const size_t base = (size_t)tid * 8;
  ((uint4*)(Ahi + base))[0] = hi.q[0];
  ((uint4*)(Ahi + base))[1] = hi.q[1];
  ((uint4*)(Alo + base))[0] = lo.q[0];
  ((uint4*)(Alo + base))[1] = lo.q[1];
}

// ---------------------------------------------------------------------------
// GEMM on pre-packed operands. One wave -> 32(M) x 64(N) tile
// (2 M-subtiles x 4 N-subtiles; B tiles reused across both M-subtiles).
// O[b, m, n] = sum_k W[m,k] * F[b,k,n] + bias[m]
// ---------------------------------------------------------------------------
__global__ __launch_bounds__(32)
void gemm_packed(const unsigned int* __restrict__ Ah,
                 const unsigned int* __restrict__ Al,
                 const unsigned int* __restrict__ Bh,
                 const unsigned int* __restrict__ Bl,
                 const float* __restrict__ bias,
                 float* __restrict__ O,
                 int M, int MT)
{
  const int lane    = threadIdx.x;
  const int l15     = lane & 15;
  const int halfSel = lane >> 4;

  const int tileN = blockIdx.x;      // covers 4 column tiles (64 cols)
  const int mt0   = blockIdx.y * 2;  // covers 2 row tiles (32 rows)
  const int b     = blockIdx.z;
  const int nt0   = tileN * 4;

  // per-lane operand bases (u32 units); mt clamped for loads, stores guarded
  const unsigned int* __restrict__ ApH[2];
  const unsigned int* __restrict__ ApL[2];
  #pragma unroll
  for (int i = 0; i < 2; ++i) {
    const int mtc = (mt0 + i) < MT ? (mt0 + i) : (MT - 1);
    ApH[i] = Ah + ((size_t)mtc * KT * 32 + lane) * 8;
    ApL[i] = Al + ((size_t)mtc * KT * 32 + lane) * 8;
  }
  const unsigned int* __restrict__ BpH = Bh + ((size_t)b * KT * NT * 32 + lane) * 8;
  const unsigned int* __restrict__ BpL = Bl + ((size_t)b * KT * NT * 32 + lane) * 8;

  v8f zero = {};
  v8f acc[2][4];
  #pragma unroll
  for (int i = 0; i < 2; ++i)
    #pragma unroll
    for (int j = 0; j < 4; ++j) acc[i][j] = zero;

  int ntc[4];
  #pragma unroll
  for (int j = 0; j < 4; ++j) {
    int nt = nt0 + j;
    ntc[j] = nt < NT ? nt : (NT - 1);   // clamp loads; stores guarded by col
  }

  for (int kt = 0; kt < KT; ++kt) {
    U8 ah[2], al[2];
    #pragma unroll
    for (int i = 0; i < 2; ++i) {
      const uint4* pah = (const uint4*)(ApH[i] + (size_t)kt * 256);
      const uint4* pal = (const uint4*)(ApL[i] + (size_t)kt * 256);
      ah[i].q[0] = pah[0]; ah[i].q[1] = pah[1];
      al[i].q[0] = pal[0]; al[i].q[1] = pal[1];
    }

    #pragma unroll
    for (int j = 0; j < 4; ++j) {
      const size_t off = ((size_t)kt * NT + ntc[j]) * 256;
      U8 bh, bl;
      const uint4* pbh = (const uint4*)(BpH + off);
      const uint4* pbl = (const uint4*)(BpL + off);
      bh.q[0] = pbh[0]; bh.q[1] = pbh[1];
      bl.q[0] = pbl[0]; bl.q[1] = pbl[1];

      #pragma unroll
      for (int i = 0; i < 2; ++i) {
        acc[i][j] = __builtin_amdgcn_wmma_f32_16x16x32_bf16(
            false, ah[i].v, false, bl.v, (short)0, acc[i][j], false, false);
        acc[i][j] = __builtin_amdgcn_wmma_f32_16x16x32_bf16(
            false, al[i].v, false, bh.v, (short)0, acc[i][j], false, false);
        acc[i][j] = __builtin_amdgcn_wmma_f32_16x16x32_bf16(
            false, ah[i].v, false, bh.v, (short)0, acc[i][j], false, false);
      }
    }
  }

  // bias per row (C/D layout: VGPR r -> M = mt*16 + r + 8*halfSel, N = l15)
  float bv[2][8];
  #pragma unroll
  for (int i = 0; i < 2; ++i)
    #pragma unroll
    for (int r = 0; r < 8; ++r) {
      const int row = (mt0 + i) * 16 + r + halfSel * 8;
      bv[i][r] = (row < M) ? bias[row] : 0.0f;
    }

  #pragma unroll
  for (int j = 0; j < 4; ++j) {
    const int col = (nt0 + j) * 16 + l15;
    if (col >= NHW) continue;
    #pragma unroll
    for (int i = 0; i < 2; ++i) {
      #pragma unroll
      for (int r = 0; r < 8; ++r) {
        const int row = (mt0 + i) * 16 + r + halfSel * 8;
        if (row < M) {
          O[((size_t)b * M + row) * NHW + col] = acc[i][j][r] + bv[i][r];
        }
      }
    }
  }
}

// ---------------------------------------------------------------------------
// Position-sensitive RoI pooling: one thread per output element.
// conv: [B, out_dim*49, 50, 84]; rois: [N,5]; out: [N, out_dim, 7, 7]
// ---------------------------------------------------------------------------
__global__ void psroi_kernel(const float* __restrict__ conv,
                             const float* __restrict__ rois,
                             float* __restrict__ out,
                             int out_dim, int total)
{
  int idx = blockIdx.x * blockDim.x + threadIdx.x;
  if (idx >= total) return;

  int pw = idx % PK;
  int t  = idx / PK;
  int ph = t % PK;  t /= PK;
  int c  = t % out_dim;
  int n  = t / out_dim;

  const float* r = rois + (size_t)n * 5;
  const int b = (int)r[0];
  const float scale = 1.0f / 16.0f;
  // jnp.round == round-half-to-even == rintf (default RNE mode)
  float sw = rintf(r[1]) * scale;
  float sh = rintf(r[2]) * scale;
  float ew = rintf(r[3] + 1.0f) * scale;
  float eh = rintf(r[4] + 1.0f) * scale;
  float bh = fmaxf(eh - sh, 0.1f) / (float)PK;
  float bw = fmaxf(ew - sw, 0.1f) / (float)PK;

  int hs = (int)fminf(fmaxf(floorf((float)ph * bh + sh), 0.0f), (float)FH);
  int he = (int)fminf(fmaxf(ceilf(((float)ph + 1.0f) * bh + sh), 0.0f), (float)FH);
  int ws = (int)fminf(fmaxf(floorf((float)pw * bw + sw), 0.0f), (float)FW);
  int we = (int)fminf(fmaxf(ceilf(((float)pw + 1.0f) * bw + sw), 0.0f), (float)FW);

  const int cnt = (he - hs) * (we - ws);
  const int o   = (c * PK + ph) * PK + pw;
  const float* __restrict__ base =
      conv + ((size_t)b * out_dim * (PK * PK) + o) * (size_t)(FH * FW);

  float sum = 0.0f;
  for (int h = hs; h < he; ++h) {
    const float* rowp = base + h * FW;
    for (int w = ws; w < we; ++w) sum += rowp[w];
  }
  out[idx] = (cnt > 0) ? (sum / (float)cnt) : 0.0f;
}

extern "C" void kernel_launch(void* const* d_in, const int* in_sizes, int n_in,
                              void* d_out, int out_size, void* d_ws, size_t ws_size,
                              hipStream_t stream) {
  const float* feats = (const float*)d_in[0];  // [2, 512, 50, 84]
  const float* rois  = (const float*)d_in[1];  // [N, 5]
  const float* w_cls = (const float*)d_in[2];  // [1029, 512]
  const float* b_cls = (const float*)d_in[3];  // [1029]
  const float* w_loc = (const float*)d_in[4];  // [4116, 512]
  const float* b_loc = (const float*)d_in[5];  // [4116]
  float* out = (float*)d_out;

  const int B    = 2;
  const int Mcls = in_sizes[3];                // 1029 = 7*7*21
  const int Mloc = in_sizes[5];                // 4116 = 7*7*84
  const int Nroi = in_sizes[1] / 5;            // 512
  const int MTc  = (Mcls + 15) / 16;           // 65
  const int MTl  = (Mloc + 15) / 16;           // 258

  // ---- workspace carve-up -------------------------------------------------
  char* w = (char*)d_ws;
  float* cls_feat = (float*)w;  w += (size_t)B * Mcls * NHW * sizeof(float);
  float* loc_feat = (float*)w;  w += (size_t)B * Mloc * NHW * sizeof(float);
  const size_t bPackN = (size_t)B * KT * NT * 32 * 8;     // u32 count
  unsigned int* Bhi = (unsigned int*)w;  w += bPackN * 4;
  unsigned int* Blo = (unsigned int*)w;  w += bPackN * 4;
  const size_t aPackC = (size_t)MTc * KT * 32 * 8;
  unsigned int* AhiC = (unsigned int*)w; w += aPackC * 4;
  unsigned int* AloC = (unsigned int*)w; w += aPackC * 4;
  const size_t aPackL = (size_t)MTl * KT * 32 * 8;
  unsigned int* AhiL = (unsigned int*)w; w += aPackL * 4;
  unsigned int* AloL = (unsigned int*)w; w += aPackL * 4;

  // ---- stage 0: pack ------------------------------------------------------
  {
    const int tot = B * KT * NT * 32;
    pack_B<<<(tot + 255) / 256, 256, 0, stream>>>(feats, Bhi, Blo);
  }
  {
    const int tot = MTc * KT * 32;
    pack_A<<<(tot + 255) / 256, 256, 0, stream>>>(w_cls, AhiC, AloC, Mcls, MTc);
  }
  {
    const int tot = MTl * KT * 32;
    pack_A<<<(tot + 255) / 256, 256, 0, stream>>>(w_loc, AhiL, AloL, Mloc, MTl);
  }

  // ---- stage 1: GEMMs -----------------------------------------------------
  dim3 blk(32);
  dim3 g_cls((NT + 3) / 4, (MTc + 1) / 2, B);
  gemm_packed<<<g_cls, blk, 0, stream>>>(AhiC, AloC, Bhi, Blo, b_cls, cls_feat,
                                         Mcls, MTc);
  dim3 g_loc((NT + 3) / 4, (MTl + 1) / 2, B);
  gemm_packed<<<g_loc, blk, 0, stream>>>(AhiL, AloL, Bhi, Blo, b_loc, loc_feat,
                                         Mloc, MTl);

  // ---- stage 2: PSROI pooling --------------------------------------------
  const int tot_cls = Nroi * 21 * PK * PK;
  psroi_kernel<<<(tot_cls + 255) / 256, 256, 0, stream>>>(
      cls_feat, rois, out, 21, tot_cls);
  const int tot_loc = Nroi * 84 * PK * PK;
  psroi_kernel<<<(tot_loc + 255) / 256, 256, 0, stream>>>(
      loc_feat, rois, out + tot_cls, 84, tot_loc);
}
